// VectorQuantizer_88725434401235
// MI455X (gfx1250) — matrix operand order repaired and checked by reading, not compiled, not run on hardware
//
#include <hip/hip_runtime.h>

typedef __attribute__((ext_vector_type(16))) __bf16 v16bf;
typedef __attribute__((ext_vector_type(4)))  __bf16 v4bf;
typedef __attribute__((ext_vector_type(8)))  float  v8f;

#define DIMS 128
#define CHUNK 64            // codes staged in LDS per iteration
#define LDS_STRIDE 144      // bf16 elements per row (288B, 32B-aligned)

// ---------------------------------------------------------------------------
// Kernel 0: per-code squared L2 norms (tiny).
// ---------------------------------------------------------------------------
__global__ void vq_prep_kernel(const float* __restrict__ emb,
                               float* __restrict__ enorm, int K)
{
    int t = blockIdx.x * blockDim.x + threadIdx.x;
    if (t >= K) return;
    const float* row = emb + (size_t)t * DIMS;
    float acc = 0.0f;
    for (int d = 0; d < DIMS; ++d) acc = fmaf(row[d], row[d], acc);
    enorm[t] = acc;
}

// ---------------------------------------------------------------------------
// Kernel 1: main VQ kernel.
//   128 threads = 4 waves; 32 rows per wave (two 16-row A tiles) -> 128 rows
//   per block.  Codebook staged through LDS in 64-code chunks, converted
//   f32 -> split bf16 (hi/lo) on the fly.  Distances via 3-term bf16 WMMA.
// ---------------------------------------------------------------------------
__global__ __launch_bounds__(128)
void vq_main_kernel(const float* __restrict__ x,
                    const float* __restrict__ emb,
                    const float* __restrict__ enorm,
                    float* __restrict__ outq,
                    float* __restrict__ outidx,
                    float* __restrict__ partials,
                    int K)
{
    __shared__ __bf16 lds_hi[CHUNK * LDS_STRIDE];   // 18 KB
    __shared__ __bf16 lds_lo[CHUNK * LDS_STRIDE];   // 18 KB
    __shared__ int    s_idx[4][32];
    __shared__ float  s_sq[4];

    const int tid  = threadIdx.x;
    const int lane = tid & 31;
    const int wid  = tid >> 5;
    const int rowbase = (blockIdx.x * 4 + wid) * 32;
    const int r    = lane & 15;   // A row-within-tile / B,C column
    const int hsel = lane >> 4;

    // ---- Build A fragments (hi/lo) for 2 row-tiles x 4 K-chunks ----------
    // bf16 A, contiguous k order used here (B matches): lanes 0-15 = rows, K = hsel*16..+15
    v16bf ah[2][4], al[2][4];
    #pragma unroll
    for (int g = 0; g < 2; ++g) {
        const float* xrow = x + (size_t)(rowbase + g * 16 + r) * DIMS;
        #pragma unroll
        for (int kk = 0; kk < 4; ++kk) {
            const float* p = xrow + kk * 32 + hsel * 16;
            float f[16];
            #pragma unroll
            for (int j = 0; j < 8; ++j) { f[j] = p[j]; f[8 + j] = p[8 + j]; }
            #pragma unroll
            for (int j = 0; j < 16; ++j) {
                __bf16 h = (__bf16)f[j];
                ah[g][kk][j] = h;
                al[g][kk][j] = (__bf16)(f[j] - (float)h);
            }
        }
    }

    float bestv[2][8];
    int   besti[2][8];
    #pragma unroll
    for (int g = 0; g < 2; ++g)
        #pragma unroll
        for (int v = 0; v < 8; ++v) { bestv[g][v] = 3.4e38f; besti[g][v] = 0; }

    const int nchunks = K / CHUNK;
    for (int ch = 0; ch < nchunks; ++ch) {
        // ---- Stage CHUNK codebook rows into LDS as split bf16 ------------
        __syncthreads();
        {
            // CHUNK*DIMS/4 = 2048 float4 / 128 threads = 16 iterations
            #pragma unroll
            for (int it = 0; it < (CHUNK * DIMS / 4) / 128; ++it) {
                const int j   = it * 128 + tid;       // float4 index
                const int row = j >> 5;               // 32 float4 per row
                const int col = (j & 31) * 4;
                const float4 f = *(const float4*)(emb +
                        (size_t)(ch * CHUNK + row) * DIMS + col);
                v4bf h, l;
                h[0] = (__bf16)f.x; l[0] = (__bf16)(f.x - (float)h[0]);
                h[1] = (__bf16)f.y; l[1] = (__bf16)(f.y - (float)h[1]);
                h[2] = (__bf16)f.z; l[2] = (__bf16)(f.z - (float)h[2]);
                h[3] = (__bf16)f.w; l[3] = (__bf16)(f.w - (float)h[3]);
                *(v4bf*)(lds_hi + row * LDS_STRIDE + col) = h;
                *(v4bf*)(lds_lo + row * LDS_STRIDE + col) = l;
            }
        }
        __syncthreads();

        // ---- 4 code tiles of 16 from this chunk --------------------------
        for (int tl = 0; tl < CHUNK / 16; ++tl) {
            const int code_local = tl * 16 + r;          // B column
            const int code       = ch * CHUNK + code_local;
            v8f acc0 = {}, acc1 = {};
            #pragma unroll
            for (int kk = 0; kk < 4; ++kk) {
                // bf16 B, same contiguous k order as A: lane holds col (lane&15),
                // K = hsel*16..+15 -> 16 contiguous bf16 (32B aligned in LDS)
                const __bf16* bb = lds_hi + code_local * LDS_STRIDE
                                 + kk * 32 + hsel * 16;
                const __bf16* ll = lds_lo + code_local * LDS_STRIDE
                                 + kk * 32 + hsel * 16;
                v16bf bh = *(const v16bf*)bb;
                v16bf bl = *(const v16bf*)ll;
                acc0 = __builtin_amdgcn_wmma_f32_16x16x32_bf16(false, ah[0][kk],
                        false, bh, (short)0, acc0, false, false);
                acc1 = __builtin_amdgcn_wmma_f32_16x16x32_bf16(false, ah[1][kk],
                        false, bh, (short)0, acc1, false, false);
                acc0 = __builtin_amdgcn_wmma_f32_16x16x32_bf16(false, ah[0][kk],
                        false, bl, (short)0, acc0, false, false);
                acc1 = __builtin_amdgcn_wmma_f32_16x16x32_bf16(false, ah[1][kk],
                        false, bl, (short)0, acc1, false, false);
                acc0 = __builtin_amdgcn_wmma_f32_16x16x32_bf16(false, al[0][kk],
                        false, bh, (short)0, acc0, false, false);
                acc1 = __builtin_amdgcn_wmma_f32_16x16x32_bf16(false, al[1][kk],
                        false, bh, (short)0, acc1, false, false);
            }
            const float en = enorm[code];
            #pragma unroll
            for (int v = 0; v < 8; ++v) {
                // ||x||^2 is row-constant -> irrelevant for argmin
                float s0 = fmaf(-2.0f, acc0[v], en);
                float s1 = fmaf(-2.0f, acc1[v], en);
                if (s0 < bestv[0][v]) { bestv[0][v] = s0; besti[0][v] = code; }
                if (s1 < bestv[1][v]) { bestv[1][v] = s1; besti[1][v] = code; }
            }
        }
    }

    // ---- Lexicographic argmin reduce across each 16-lane half -------------
    #pragma unroll
    for (int g = 0; g < 2; ++g)
        #pragma unroll
        for (int v = 0; v < 8; ++v)
            #pragma unroll
            for (int off = 8; off >= 1; off >>= 1) {
                float ov = __shfl_xor(bestv[g][v], off, 16);
                int   oi = __shfl_xor(besti[g][v], off, 16);
                if (ov < bestv[g][v] ||
                    (ov == bestv[g][v] && oi < besti[g][v])) {
                    bestv[g][v] = ov; besti[g][v] = oi;
                }
            }
    // C/D layout: VGPR v holds row v (lanes 0-15) / row v+8 (lanes 16-31)
    if (r == 0) {
        #pragma unroll
        for (int g = 0; g < 2; ++g)
            #pragma unroll
            for (int v = 0; v < 8; ++v)
                s_idx[wid][g * 16 + hsel * 8 + v] = besti[g][v];
    }
    __syncthreads();

    // ---- Emit indices (as float, per harness output dtype) ---------------
    outidx[rowbase + lane] = (float)s_idx[wid][lane];

    // ---- Gather exact f32 codebook rows -> quantized; exact loss accum ----
    float sq = 0.0f;
    for (int rr = 0; rr < 32; ++rr) {
        const int idx = s_idx[wid][rr];
        const float4 ev = *(const float4*)(emb + (size_t)idx * DIMS + lane * 4);
        const float4 xv = *(const float4*)(x +
                (size_t)(rowbase + rr) * DIMS + lane * 4);
        *(float4*)(outq + (size_t)(rowbase + rr) * DIMS + lane * 4) = ev;
        float dx = ev.x - xv.x, dy = ev.y - xv.y;
        float dz = ev.z - xv.z, dw = ev.w - xv.w;
        sq += dx * dx + dy * dy + dz * dz + dw * dw;
    }
    #pragma unroll
    for (int off = 16; off >= 1; off >>= 1) sq += __shfl_xor(sq, off, 32);
    if (lane == 0) s_sq[wid] = sq;
    __syncthreads();
    if (tid == 0)
        partials[blockIdx.x] = s_sq[0] + s_sq[1] + s_sq[2] + s_sq[3];
}

// ---------------------------------------------------------------------------
// Kernel 2: deterministic loss reduction.
// loss = (1 + 0.25) * mean((q - x)^2)  [stop-grad makes both terms identical]
// ---------------------------------------------------------------------------
__global__ void vq_loss_reduce_kernel(const float* __restrict__ partials, int n,
                                      float* __restrict__ loss, float scale)
{
    __shared__ float s[256];
    float a = 0.0f;
    for (int i = threadIdx.x; i < n; i += 256) a += partials[i];
    s[threadIdx.x] = a;
    __syncthreads();
    for (int o = 128; o > 0; o >>= 1) {
        if (threadIdx.x < o) s[threadIdx.x] += s[threadIdx.x + o];
        __syncthreads();
    }
    if (threadIdx.x == 0) loss[0] = s[0] * scale;
}

// ---------------------------------------------------------------------------
extern "C" void kernel_launch(void* const* d_in, const int* in_sizes, int n_in,
                              void* d_out, int out_size, void* d_ws, size_t ws_size,
                              hipStream_t stream)
{
    const float* x   = (const float*)d_in[0];   // [N, 128] f32
    const float* emb = (const float*)d_in[1];   // [K, 128] f32

    const int N = in_sizes[0] / DIMS;           // 65536
    const int K = in_sizes[1] / DIMS;           // 1024
    const int numBlocks = N / 128;              // 4 waves * 32 rows per block

    // workspace: enorm (K f32) + block partials
    float* enorm = (float*)d_ws;
    float* parts = enorm + K;

    // output layout: [loss(1) | quantized(N*128) | indices(N)]
    float* loss   = (float*)d_out;
    float* outq   = loss + 1;
    float* outidx = outq + (size_t)N * DIMS;

    vq_prep_kernel<<<(K + 255) / 256, 256, 0, stream>>>(emb, enorm, K);

    vq_main_kernel<<<numBlocks, 128, 0, stream>>>(x, emb, enorm,
                                                  outq, outidx, parts, K);

    const float scale = 1.25f / (float)((size_t)N * DIMS);
    vq_loss_reduce_kernel<<<1, 256, 0, stream>>>(parts, numBlocks, loss, scale);
}
